// MultiHeadAttentionLayer_33595234189451
// MI455X (gfx1250) — compile-verified, matrix operand order checked by reading, not run on hardware
//
#include <hip/hip_runtime.h>
#include <hip/hip_bf16.h>

#define HNUM 8
#define BB 8
#define TT 1024
#define HH 2048
#define DH 256
#define H3 6144
#define BT 8192   // B*T

typedef __attribute__((ext_vector_type(16))) __bf16 bf16x16;
typedef __attribute__((ext_vector_type(8)))  float  f32x8;

union FragU { uint4 u[2]; bf16x16 v; };

// Load a 16x32 bf16 WMMA operand fragment (A: outer=M base; B: outer=N base of W^T).
// CDNA5 16-bit operand layout: lanes 0-15 hold rows 0-15 with K {0..7,16..23},
// lanes 16-31 hold rows 0-15 with K {8..15,24..31}.
__device__ __forceinline__ bf16x16 load_frag(const __hip_bfloat16* base, int ld, int outer, int k0) {
  int lane = threadIdx.x & 31;
  int r  = outer + (lane & 15);
  int kb = k0 + ((lane >> 4) << 3);
  const uint4* p = (const uint4*)(base + (size_t)r * ld + kb);
  FragU f;
  f.u[0] = p[0];
  f.u[1] = p[2];   // +32 bytes = +16 halves
  return f.v;
}

__device__ __forceinline__ f32x8 wmma_bf16(bf16x16 a, bf16x16 b, f32x8 c) {
  return __builtin_amdgcn_wmma_f32_16x16x32_bf16(false, a, false, b, (short)0, c, false, false);
}

// Async 16B copy global -> LDS (CDNA5 GLOBAL_LOAD_ASYNC_TO_LDS_B128, tracked by ASYNCcnt).
__device__ __forceinline__ void async_copy16(unsigned lds_off, const __hip_bfloat16* gaddr) {
  asm volatile("global_load_async_to_lds_b128 %0, %1, off"
               :: "v"(lds_off), "v"((unsigned long long)(uintptr_t)gaddr) : "memory");
}

// ---------------- prep: transpose + cast weights to bf16 ----------------
__global__ void prep_w_kernel(const float* __restrict__ qkv_w, const float* __restrict__ o_w,
                              __hip_bfloat16* __restrict__ qkvwT, __hip_bfloat16* __restrict__ owT) {
  const size_t N1 = (size_t)HH * H3;
  const size_t N2 = (size_t)HH * HH;
  for (size_t i = (size_t)blockIdx.x * blockDim.x + threadIdx.x; i < N1 + N2;
       i += (size_t)gridDim.x * blockDim.x) {
    if (i < N1) {
      size_t k = i / H3, c = i - k * H3;
      qkvwT[c * HH + k] = __float2bfloat16(qkv_w[i]);
    } else {
      size_t j = i - N1;
      size_t k = j / HH, c = j - k * HH;
      owT[c * HH + k] = __float2bfloat16(o_w[j]);
    }
  }
}

// ---------------- LayerNorm -> bf16 ----------------
__global__ void ln_bf16_kernel(const float* __restrict__ x, const float* __restrict__ w,
                               const float* __restrict__ bsrc, __hip_bfloat16* __restrict__ out) {
  __shared__ float red[256];
  int row = blockIdx.x;
  int tid = threadIdx.x;
  const float* xr = x + (size_t)row * HH;
  float v[8];
  float s = 0.f;
#pragma unroll
  for (int i = 0; i < 8; i++) { v[i] = xr[tid + i * 256]; s += v[i]; }
  red[tid] = s; __syncthreads();
  for (int off = 128; off > 0; off >>= 1) {
    if (tid < off) red[tid] += red[tid + off];
    __syncthreads();
  }
  float mu = red[0] * (1.f / HH);
  __syncthreads();
  float s2 = 0.f;
#pragma unroll
  for (int i = 0; i < 8; i++) { float d = v[i] - mu; s2 += d * d; }
  red[tid] = s2; __syncthreads();
  for (int off = 128; off > 0; off >>= 1) {
    if (tid < off) red[tid] += red[tid + off];
    __syncthreads();
  }
  float rstd = rsqrtf(red[0] * (1.f / HH) + 1e-8f);
  __hip_bfloat16* orow = out + (size_t)row * HH;
#pragma unroll
  for (int i = 0; i < 8; i++) {
    int c = tid + i * 256;
    orow[c] = __float2bfloat16((v[i] - mu) * rstd * w[c] + bsrc[c]);
  }
}

// ============ shared GEMM core: block = M128 x N64, B tile async-staged in LDS ============
// bstage: double buffer, each 64 rows (N cols of W^T) x 32 K halves = 4KB.
// Copy pattern: thread t -> row t/4, 16B chunk t%4.
struct GemmAcc { f32x8 a[4]; };

__device__ __forceinline__ GemmAcc gemm_core(const __hip_bfloat16* __restrict__ a_src,
                                             const __hip_bfloat16* __restrict__ wT_src,
                                             int m0, int n0,
                                             __hip_bfloat16* bstage /* shared, 2*2048 halves */) {
  int tid = threadIdx.x;
  int crow = tid >> 2;          // 0..63
  int cchk = tid & 3;           // 16B chunk
  const __hip_bfloat16* gsrc = wT_src + (size_t)(n0 + crow) * HH + (cchk << 3);
  unsigned lds0 = (unsigned)(uintptr_t)bstage + (crow << 6) + (cchk << 4);
  unsigned lds1 = lds0 + 4096;

  f32x8 z = {0.f, 0.f, 0.f, 0.f, 0.f, 0.f, 0.f, 0.f};
  GemmAcc acc;
#pragma unroll
  for (int i = 0; i < 4; i++) acc.a[i] = z;

  // prologue: stage k-step 0, preload A frag 0
  async_copy16(lds0, gsrc);
  bf16x16 a = load_frag(a_src, HH, m0, 0);

  for (int it = 0; it < HH / 32; ++it) {
    int k0 = it << 5;
    if (it + 1 < HH / 32) {
      async_copy16((it & 1) ? lds0 : lds1, gsrc + k0 + 32);
      asm volatile("s_wait_asynccnt 0x1" ::: "memory");   // buffer for THIS step landed
    } else {
      asm volatile("s_wait_asynccnt 0x0" ::: "memory");
    }
    __syncthreads();                                      // all waves' chunks visible

    bf16x16 a_next = a;
    if (it + 1 < HH / 32) {
      a_next = load_frag(a_src, HH, m0, k0 + 32);
      __builtin_prefetch(a_src + (size_t)m0 * HH + k0 + 64, 0, 3);
    }
    const __hip_bfloat16* bbuf = bstage + ((it & 1) << 11);   // buffer it&1

    // batch all 8 ds_load_b128 first, then issue the 4 WMMAs back-to-back
    bf16x16 bf[4];
#pragma unroll
    for (int s = 0; s < 4; s++) bf[s] = load_frag(bbuf, 32, s << 4, 0);
#pragma unroll
    for (int s = 0; s < 4; s++) acc.a[s] = wmma_bf16(a, bf[s], acc.a[s]);

    a = a_next;
    __syncthreads();                                      // reads done before next overwrite
  }
  return acc;
}

// ---------------- QKV GEMM: [8192x2048] @ [2048x6144] -> scatter q/k/vT ----------------
__global__ void qkv_gemm_kernel(const __hip_bfloat16* __restrict__ xln,
                                const __hip_bfloat16* __restrict__ wT,
                                __hip_bfloat16* __restrict__ qb,
                                __hip_bfloat16* __restrict__ kbuf,
                                __hip_bfloat16* __restrict__ vT) {
  __shared__ __align__(16) __hip_bfloat16 bstage[2 * 2048];
  const int NB = H3 / 64;                 // 96 N-blocks
  int mb = blockIdx.x / NB;
  int nb = blockIdx.x - mb * NB;
  int wid = threadIdx.x >> 5;
  int lane = threadIdx.x & 31;
  int m0 = (mb << 7) + (wid << 4);
  int n0 = nb << 6;

  GemmAcc acc = gemm_core(xln, wT, m0, n0, bstage);

  int hi8 = (lane >> 4) << 3;
#pragma unroll
  for (int s = 0; s < 4; s++) {
    int c = n0 + (s << 4) + (lane & 15);
    int which = c / HH;
    int hc = c - which * HH;
    int head = hc >> 8;
    int d = hc & 255;
#pragma unroll
    for (int j = 0; j < 8; j++) {
      int m = m0 + j + hi8;
      int b = m >> 10, t = m & 1023;
      __hip_bfloat16 val = __float2bfloat16(acc.a[s][j]);
      size_t bh = (size_t)b * HNUM + head;
      if (which == 0)      qb  [(bh * TT + t) * DH + d] = val;
      else if (which == 1) kbuf[(bh * TT + t) * DH + d] = val;
      else                 vT  [(bh * DH + d) * TT + t] = val;
    }
  }
}

// ---------------- out GEMM + residual ----------------
__global__ void out_gemm_kernel(const __hip_bfloat16* __restrict__ ctx,
                                const __hip_bfloat16* __restrict__ owT,
                                const float* __restrict__ inp,
                                float* __restrict__ out) {
  __shared__ __align__(16) __hip_bfloat16 bstage[2 * 2048];
  const int NB = HH / 64;                 // 32 N-blocks
  int mb = blockIdx.x / NB;
  int nb = blockIdx.x - mb * NB;
  int wid = threadIdx.x >> 5;
  int lane = threadIdx.x & 31;
  int m0 = (mb << 7) + (wid << 4);
  int n0 = nb << 6;

  GemmAcc acc = gemm_core(ctx, owT, m0, n0, bstage);

  int hi8 = (lane >> 4) << 3;
#pragma unroll
  for (int s = 0; s < 4; s++) {
    int c = n0 + (s << 4) + (lane & 15);
#pragma unroll
    for (int j = 0; j < 8; j++) {
      size_t m = (size_t)(m0 + j + hi8);
      out[m * HH + c] = acc.a[s][j] + inp[m * HH + c];
    }
  }
}

// ---------------- flash attention: one wave = 16 queries x full dh for one (b,h) ----------------
__global__ void attn_kernel(const __hip_bfloat16* __restrict__ qb,
                            const __hip_bfloat16* __restrict__ kbuf,
                            const __hip_bfloat16* __restrict__ vT,
                            const int* __restrict__ mask,
                            const int* __restrict__ isPre_p,
                            __hip_bfloat16* __restrict__ ctx) {
  __shared__ __align__(16) __hip_bfloat16 pbuf[8][16][32];
  int wid  = threadIdx.x >> 5;
  int lane = threadIdx.x & 31;
  int wave = (blockIdx.x << 3) + wid;
  int b   = wave >> 9;
  int rem = wave & 511;
  int h   = rem >> 6;
  int qt  = rem & 63;
  int q0  = qt << 4;
  int isPre = isPre_p[0];

  const __hip_bfloat16* qbh = qb   + ((size_t)b * HNUM + h) * TT * DH;
  const __hip_bfloat16* kbh = kbuf + ((size_t)b * HNUM + h) * TT * DH;
  const __hip_bfloat16* vbh = vT   + ((size_t)b * HNUM + h) * DH * TT;
  const int* mrow = mask + b * TT;

  bf16x16 qf[8];
#pragma unroll
  for (int kk = 0; kk < 8; kk++) qf[kk] = load_frag(qbh, DH, q0, kk * 32);

  f32x8 z = {0.f, 0.f, 0.f, 0.f, 0.f, 0.f, 0.f, 0.f};
  f32x8 o[16];
#pragma unroll
  for (int n = 0; n < 16; n++) o[n] = z;
  float m[8], l[8];
#pragma unroll
  for (int j = 0; j < 8; j++) { m[j] = -INFINITY; l[j] = 0.f; }
  int hi8 = (lane >> 4) << 3;
  int nlo = lane & 15;

  for (int kb = 0; kb < TT; kb += 32) {
    f32x8 s0 = z, s1 = z;
    // software-pipelined K-fragment loads: operands for step kk+1 in flight during step kk
    bf16x16 b0 = load_frag(kbh, DH, kb,      0);
    bf16x16 b1 = load_frag(kbh, DH, kb + 16, 0);
#pragma unroll
    for (int kk = 0; kk < 8; kk++) {
      bf16x16 n0f = b0, n1f = b1;
      if (kk < 7) {
        n0f = load_frag(kbh, DH, kb,      (kk + 1) * 32);
        n1f = load_frag(kbh, DH, kb + 16, (kk + 1) * 32);
      }
      s0 = wmma_bf16(qf[kk], b0, s0);
      s1 = wmma_bf16(qf[kk], b1, s1);
      b0 = n0f; b1 = n1f;
    }
    int kc0 = kb + nlo, kc1 = kc0 + 16;
    int mk0 = mrow[kc0], mk1 = mrow[kc1];
    float corr[8];
#pragma unroll
    for (int j = 0; j < 8; j++) {
      int qrow = q0 + j + hi8;
      float v0 = s0[j], v1 = s1[j];
      if (mk0 || (isPre && kc0 > qrow)) v0 = -1e9f;   // mask BEFORE scale (faithful to ref)
      if (mk1 || (isPre && kc1 > qrow)) v1 = -1e9f;
      v0 *= 0.0625f; v1 *= 0.0625f;                   // 1/sqrt(256)
      float mx = fmaxf(v0, v1);
#pragma unroll
      for (int off = 1; off < 16; off <<= 1) mx = fmaxf(mx, __shfl_xor(mx, off, 32));
      float mnew = fmaxf(m[j], mx);
      float cf = __expf(m[j] - mnew);
      float p0 = __expf(v0 - mnew);
      float p1 = __expf(v1 - mnew);
      float ps = p0 + p1;
#pragma unroll
      for (int off = 1; off < 16; off <<= 1) ps += __shfl_xor(ps, off, 32);
      l[j] = l[j] * cf + ps;
      m[j] = mnew;
      corr[j] = cf;
      int r = j + hi8;
      pbuf[wid][r][nlo]      = __float2bfloat16(p0);
      pbuf[wid][r][nlo + 16] = __float2bfloat16(p1);
    }
    asm volatile("s_wait_dscnt 0" ::: "memory");   // LDS RAW: P transpose staging
    bf16x16 pf = load_frag(&pbuf[wid][0][0], 32, 0, 0);
    // software-pipelined V-fragment loads
    bf16x16 vf = load_frag(vbh, TT, 0, kb);
#pragma unroll
    for (int n = 0; n < 16; n++) {
      bf16x16 vnext = vf;
      if (n < 15) vnext = load_frag(vbh, TT, (n + 1) << 4, kb);
      f32x8 oc = o[n];
#pragma unroll
      for (int j = 0; j < 8; j++) oc[j] *= corr[j];
      o[n] = wmma_bf16(pf, vf, oc);
      vf = vnext;
    }
  }

#pragma unroll
  for (int n = 0; n < 16; n++) {
    int d = (n << 4) + nlo;
#pragma unroll
    for (int j = 0; j < 8; j++) {
      int t = q0 + j + hi8;
      float val = o[n][j] / l[j];
      ctx[((size_t)b * TT + t) * HH + h * DH + d] = __float2bfloat16(val);
    }
  }
}

extern "C" void kernel_launch(void* const* d_in, const int* in_sizes, int n_in,
                              void* d_out, int out_size, void* d_ws, size_t ws_size,
                              hipStream_t stream) {
  const float* inputs = (const float*)d_in[0];
  const int*   mask   = (const int*)d_in[1];
  const float* lnw    = (const float*)d_in[2];
  const float* lnb    = (const float*)d_in[3];
  const float* qkv_w  = (const float*)d_in[4];
  const float* o_w    = (const float*)d_in[5];
  const int*   isPre  = (const int*)d_in[6];
  float* out = (float*)d_out;

  char* p = (char*)d_ws;
  __hip_bfloat16* qkvwT = (__hip_bfloat16*)p; p += (size_t)H3 * HH * 2;
  __hip_bfloat16* owT   = (__hip_bfloat16*)p; p += (size_t)HH * HH * 2;
  __hip_bfloat16* xln   = (__hip_bfloat16*)p; p += (size_t)BT * HH * 2;
  __hip_bfloat16* qb    = (__hip_bfloat16*)p; p += (size_t)BT * HH * 2;
  __hip_bfloat16* kbuf  = (__hip_bfloat16*)p; p += (size_t)BT * HH * 2;
  __hip_bfloat16* vT    = (__hip_bfloat16*)p; p += (size_t)BT * HH * 2;
  __hip_bfloat16* ctx   = (__hip_bfloat16*)p; p += (size_t)BT * HH * 2;

  prep_w_kernel<<<4096, 256, 0, stream>>>(qkv_w, o_w, qkvwT, owT);
  ln_bf16_kernel<<<BT, 256, 0, stream>>>(inputs, lnw, lnb, xln);
  // 64 M-blocks (128 rows) x 96 N-blocks (64 cols) = 6144 blocks
  qkv_gemm_kernel<<<6144, 256, 0, stream>>>(xln, qkvwT, qb, kbuf, vT);
  // 8*8*64 = 4096 waves / 8 per block
  attn_kernel<<<512, 256, 0, stream>>>(qb, kbuf, vT, mask, isPre, ctx);
  // 64 M-blocks x 32 N-blocks = 2048 blocks
  out_gemm_kernel<<<2048, 256, 0, stream>>>(ctx, owT, inputs, out);
}
